// GraphEncoderLayer_57320633533153
// MI455X (gfx1250) — matrix operand order repaired and checked by reading, not compiled, not run on hardware
//
#include <hip/hip_runtime.h>
#include <hip/hip_bf16.h>
#include <math.h>

typedef __attribute__((ext_vector_type(16))) __bf16 v16bf;
typedef __attribute__((ext_vector_type(8)))  __bf16 v8bf;
typedef __attribute__((ext_vector_type(4)))  __bf16 v4bf;
typedef __attribute__((ext_vector_type(2)))  __bf16 v2bf;
typedef __attribute__((ext_vector_type(8)))  float  v8f;

#define DEVFN static __device__ __forceinline__

static constexpr int Bb = 8, Nn = 512, KK = 16, Dd = 256, Ff = 1024, Hh = 8;
// LDS strides (elements)
static constexpr int A_LD = 40;    // bf16 32x32 tile (rows 80B, 16B aligned)
static constexpr int AP_LD = 264;  // bf16 32x256 A panel (rows 528B, 16B aligned)
static constexpr int H_LD = 1040;  // bf16 32x1024 hidden
static constexpr int C_LD = 260;   // f32  32x256 C stage
static constexpr int P_LD = 520;   // 32x512 prob tile
static constexpr int KV_LD = 40;   // bf16 512x32 K tile
static constexpr int VT_LD = 520;  // bf16 32x512 V-transposed tile
static constexpr int BSW_PLANE = 4096;  // 16 col-tiles * 32 lanes * 8

DEVFN v8f wmma_bf16(v16bf a, v16bf b, v8f c) {
  return __builtin_amdgcn_wmma_f32_16x16x32_bf16(false, a, false, b, (short)0, c,
                                                 false, false);
}

// A fragment 16x32 bf16 from row-major LDS: two contiguous 16B runs per lane.
DEVFN v16bf load_a_frag(const __bf16* base, int ld) {
  const int lane = threadIdx.x & 31;
  const int half = lane >> 4, m = lane & 15;
  const __bf16* row = base + m * ld + 16 * half;
  const v8bf lo = *(const v8bf*)(row);
  const v8bf hi = *(const v8bf*)(row + 8);
  return __builtin_shufflevector(lo, hi, 0, 1, 2, 3, 4, 5, 6, 7,
                                 8, 9, 10, 11, 12, 13, 14, 15);
}

// B fragment from swizzled tile: plane layout [jt][lane][8], two planes.
DEVFN v16bf load_b_frag_sw(const __bf16* Bsw, int jt) {
  const int lane = threadIdx.x & 31;
  const __bf16* p = Bsw + (jt * 32 + lane) * 8;
  const v8bf lo = *(const v8bf*)(p);
  const v8bf hi = *(const v8bf*)(p + BSW_PLANE);
  return __builtin_shufflevector(lo, hi, 0, 1, 2, 3, 4, 5, 6, 7,
                                 8, 9, 10, 11, 12, 13, 14, 15);
}

// B fragment where source is N x K row-major (B[k][n] = src[n][k]).
DEVFN v16bf load_bT_frag(const __bf16* base, int ld) {
  const int lane = threadIdx.x & 31;
  const int half = lane >> 4, n = lane & 15;
  const __bf16* row = base + n * ld + 16 * half;
  const v8bf lo = *(const v8bf*)(row);
  const v8bf hi = *(const v8bf*)(row + 8);
  return __builtin_shufflevector(lo, hi, 0, 1, 2, 3, 4, 5, 6, 7,
                                 8, 9, 10, 11, 12, 13, 14, 15);
}

// Cooperative 32x256 A-panel load (once per block), f32->bf16 or bf16 copy.
// AMODE 0: row-major leading dim lda. AMODE 1: x stored (N,B,D).
template <int AMODE, bool ABF16>
DEVFN void load_a_panel(const void* A, int rowBase, int lda, __bf16* Asp) {
  const int t = threadIdx.x;
  const int r = t >> 3, c0 = (t & 7) * 32;
  const int gr = rowBase + r;
  long off;
  if (AMODE == 1) {
    const int b = gr >> 9, n = gr & (Nn - 1);
    off = ((long)n * Bb + b) * Dd;
  } else {
    off = (long)gr * lda;
  }
  if (ABF16) {
    const v8bf* p = (const v8bf*)((const __bf16*)A + off + c0);
#pragma unroll
    for (int i = 0; i < 4; ++i) *(v8bf*)(Asp + r * AP_LD + c0 + i * 8) = p[i];
  } else {
    const float4* p4 = (const float4*)((const float*)A + off + c0);
#pragma unroll
    for (int i = 0; i < 8; ++i) {
      const float4 f = p4[i];
      v4bf pk = {(__bf16)f.x, (__bf16)f.y, (__bf16)f.z, (__bf16)f.w};
      *(v4bf*)(Asp + r * AP_LD + c0 + i * 4) = pk;
    }
  }
}

// Cooperative 32x256 weight-tile load, f32 -> bf16, swizzled fragment-major.
DEVFN void load_b_tile_sw(const float* W, int k0, int colBase, int ldw, __bf16* Bsw) {
  const int t = threadIdx.x;
  const int rg = t >> 5, cg = t & 31;
  const int r0 = rg * 4, c0 = cg * 8;
  const int half = r0 >> 4;
  const int seg = (r0 & 15) >> 3;
  const int o = r0 & 7;  // 0 or 4
  __bf16 vals[4][8];
#pragma unroll
  for (int j = 0; j < 4; ++j) {
    const float4* p4 = (const float4*)(W + (long)(k0 + r0 + j) * ldw + colBase + c0);
    const float4 f0 = p4[0], f1 = p4[1];
    vals[j][0] = (__bf16)f0.x; vals[j][1] = (__bf16)f0.y;
    vals[j][2] = (__bf16)f0.z; vals[j][3] = (__bf16)f0.w;
    vals[j][4] = (__bf16)f1.x; vals[j][5] = (__bf16)f1.y;
    vals[j][6] = (__bf16)f1.z; vals[j][7] = (__bf16)f1.w;
  }
#pragma unroll
  for (int i = 0; i < 8; ++i) {
    const int c = c0 + i, jt = c >> 4, n = c & 15;
    const int l = half * 16 + n;
    v4bf pk = {vals[0][i], vals[1][i], vals[2][i], vals[3][i]};
    *(v4bf*)(Bsw + seg * BSW_PLANE + (jt * 32 + l) * 8 + o) = pk;
  }
}

// ---------------------------------------------------------------- kernels

__global__ void zero_kernel(float* p, long n) {
  long i = ((long)blockIdx.x * blockDim.x + threadIdx.x) * 4;
  if (i + 3 < n) { p[i] = 0.f; p[i + 1] = 0.f; p[i + 2] = 0.f; p[i + 3] = 0.f; }
}

// bias[b, n, idx] = mean(e[b,n,k,:]) ; one wave per edge row
__global__ void scatter_bias_kernel(const float* __restrict__ e,
                                    const int* __restrict__ eidx,
                                    float* __restrict__ bias) {
  const int wave = threadIdx.x >> 5, lane = threadIdx.x & 31;
  const int row = blockIdx.x * 8 + wave;
  const float* er = e + (long)row * Dd;
  float s = 0.f;
#pragma unroll
  for (int i = 0; i < 8; ++i) s += er[lane + 32 * i];
#pragma unroll
  for (int m = 16; m; m >>= 1) s += __shfl_xor(s, m, 32);
  if (lane == 0) {
    const int b = row >> 13, n = (row >> 4) & (Nn - 1);
    bias[((long)(b * Nn + n)) * Nn + eidx[row]] = s * (1.0f / 256.0f);
  }
}

// (xt @ W + bvec) * scale -> bf16, scattered into (B,H,N,32) layout
__global__ void qkv_gemm_kernel(const float* __restrict__ x,
                                const float* __restrict__ W,
                                const float* __restrict__ bvec,
                                __bf16* __restrict__ dst, float scale) {
  __shared__ __bf16 Asp[32 * AP_LD];
  __shared__ __bf16 Bsw[2 * BSW_PLANE];
  const int rowBase = blockIdx.x * 32;
  const int wave = threadIdx.x >> 5, lane = threadIdx.x & 31;
  const int half = lane >> 4, nl = lane & 15;
  const int wrow = (wave >> 2) * 16, wcol = (wave & 3) * 64;
  load_a_panel<1, false>(x, rowBase, Dd, Asp);
  v8f acc[4] = {};
  for (int k0 = 0; k0 < Dd; k0 += 32) {
    load_b_tile_sw(W, k0, 0, Dd, Bsw);
    __syncthreads();
    const v16bf a = load_a_frag(Asp + wrow * AP_LD + k0, AP_LD);
#pragma unroll
    for (int j = 0; j < 4; ++j)
      acc[j] = wmma_bf16(a, load_b_frag_sw(Bsw, (wcol >> 4) + j), acc[j]);
    __syncthreads();
  }
  // epilogue: block-constant b/n0, constant stride stores
  const int b = rowBase >> 9;
  const int n0 = (rowBase & (Nn - 1)) + wrow + 8 * half;
#pragma unroll
  for (int j = 0; j < 4; ++j) {
    const int c = wcol + 16 * j + nl;
    const float bv = bvec[c];
    const int h = c >> 5, dd = c & 31;
    __bf16* p = dst + (((long)(b * Hh + h)) * Nn + n0) * 32 + dd;
#pragma unroll
    for (int r = 0; r < 8; ++r) p[r * 32] = (__bf16)((acc[j][r] + bv) * scale);
  }
}

// One WG per (b, 32-query tile). Loops 8 heads: S=Q@K^T+bias (WMMA), softmax,
// O=P@V (WMMA), accumulates mean-over-heads attention -> weights. No atomics.
__global__ void attn_kernel(const __bf16* __restrict__ qb,
                            const __bf16* __restrict__ kb,
                            const __bf16* __restrict__ vb,
                            const float* __restrict__ bias,
                            float* __restrict__ weights,
                            float* __restrict__ attn_out) {
  __shared__ __bf16 Qs[32 * A_LD];
  __shared__ __bf16 Ks[Nn * KV_LD];
  __shared__ __bf16 Vt[32 * VT_LD];     // V transposed: [d][k]
  __shared__ __bf16 Pb[32 * P_LD];
  __shared__ float  Wsum[32 * P_LD];
  const int qt = blockIdx.x & 15, b = blockIdx.x >> 4;
  const int qBase = qt * 32;
  const int wave = threadIdx.x >> 5, lane = threadIdx.x & 31;
  const int half = lane >> 4, nl = lane & 15;
  const int wrow = (wave >> 2) * 16;

  for (int i = threadIdx.x * 4; i < 32 * P_LD; i += 256 * 4)
    *(float4*)&Wsum[i] = float4{0.f, 0.f, 0.f, 0.f};

  for (int h = 0; h < Hh; ++h) {
    __syncthreads();
    const long headOff = ((long)(b * Hh + h)) * Nn * 32;
    {  // load Q tile (32x32)
      const int t = threadIdx.x, r = t >> 3, c0 = (t & 7) * 4;
      *(v4bf*)(Qs + r * A_LD + c0) =
          *(const v4bf*)(qb + headOff + (long)(qBase + r) * 32 + c0);
    }
    {  // load K (row-major) and V (transposed) tiles
      const int t = threadIdx.x;
      const int r0 = t * 2;
      const int4* pk0 = (const int4*)(kb + headOff + (long)r0 * 32);
      const int4* pk1 = (const int4*)(kb + headOff + (long)(r0 + 1) * 32);
#pragma unroll
      for (int i = 0; i < 4; ++i) {
        *(int4*)&Ks[r0 * KV_LD + i * 8] = pk0[i];
        *(int4*)&Ks[(r0 + 1) * KV_LD + i * 8] = pk1[i];
      }
      const v8bf* pv0 = (const v8bf*)(vb + headOff + (long)r0 * 32);
      const v8bf* pv1 = (const v8bf*)(vb + headOff + (long)(r0 + 1) * 32);
      __bf16* vdst = Vt + r0;
#pragma unroll
      for (int sg = 0; sg < 4; ++sg) {
        const v8bf a = pv0[sg], c = pv1[sg];
#pragma unroll
        for (int i = 0; i < 8; ++i) {
          v2bf pr = {a[i], c[i]};
          *(v2bf*)(vdst + (sg * 8 + i) * VT_LD) = pr;
        }
      }
    }
    __syncthreads();
    // S = Q @ K^T + bias  -> Pb (bf16)
    {
      const v16bf aq = load_a_frag(Qs + wrow * A_LD, A_LD);
      const int m0 = wrow + 8 * half;
      const float* brow = bias + ((long)(b * Nn + qBase + m0)) * Nn + nl;
      __bf16* prow = Pb + m0 * P_LD + nl;
      for (int jj = (wave & 3); jj < 32; jj += 4) {
        const int colBase = jj * 16;
        const v16bf bk = load_bT_frag(Ks + colBase * KV_LD, KV_LD);
        v8f s = {};
        s = wmma_bf16(aq, bk, s);
#pragma unroll
        for (int r = 0; r < 8; ++r)
          prow[r * P_LD + colBase] = (__bf16)(s[r] + brow[(long)r * Nn + colBase]);
      }
    }
    __syncthreads();
    // softmax per row (8 lanes x 64 cols per row), accumulate head mean
    {
      const int t = threadIdx.x, row = t >> 3, seg = t & 7;
      __bf16* prow = Pb + row * P_LD + seg * 64;
      float mx = -1e30f;
#pragma unroll
      for (int i = 0; i < 64; i += 4) {
        const v4bf p = *(const v4bf*)(prow + i);
        mx = fmaxf(mx, fmaxf(fmaxf((float)p[0], (float)p[1]),
                             fmaxf((float)p[2], (float)p[3])));
      }
#pragma unroll
      for (int m = 1; m < 8; m <<= 1) mx = fmaxf(mx, __shfl_xor(mx, m, 32));
      float sum = 0.f;
#pragma unroll
      for (int i = 0; i < 64; i += 4) {
        const v4bf p = *(const v4bf*)(prow + i);
        sum += __expf((float)p[0] - mx) + __expf((float)p[1] - mx) +
               __expf((float)p[2] - mx) + __expf((float)p[3] - mx);
      }
#pragma unroll
      for (int m = 1; m < 8; m <<= 1) sum += __shfl_xor(sum, m, 32);
      const float inv = 1.0f / sum;
      float* wrow_s = Wsum + row * P_LD + seg * 64;
#pragma unroll
      for (int i = 0; i < 64; i += 4) {
        const v4bf p = *(const v4bf*)(prow + i);
        const float p0 = __expf((float)p[0] - mx) * inv;
        const float p1 = __expf((float)p[1] - mx) * inv;
        const float p2 = __expf((float)p[2] - mx) * inv;
        const float p3 = __expf((float)p[3] - mx) * inv;
        v4bf o = {(__bf16)p0, (__bf16)p1, (__bf16)p2, (__bf16)p3};
        *(v4bf*)(prow + i) = o;
        float4 wv = *(float4*)(wrow_s + i);
        wv.x += p0 * 0.125f; wv.y += p1 * 0.125f;
        wv.z += p2 * 0.125f; wv.w += p3 * 0.125f;
        *(float4*)(wrow_s + i) = wv;
      }
    }
    __syncthreads();
    // O = P @ V  (waves 0..3: 2 row-tiles x 2 col-tiles)
    if (wave < 4) {
      const int rowt = (wave >> 1) * 16, colt = (wave & 1) * 16;
      v8f o = {};
      for (int k0 = 0; k0 < Nn; k0 += 32) {
        const v16bf ap = load_a_frag(Pb + rowt * P_LD + k0, P_LD);
        const v16bf bv = load_bT_frag(Vt + colt * VT_LD + k0, VT_LD);
        o = wmma_bf16(ap, bv, o);
      }
      float* op = attn_out + ((long)(b * Nn + qBase + rowt + 8 * half)) * Dd +
                  h * 32 + colt + nl;
#pragma unroll
      for (int r = 0; r < 8; ++r) op[(long)r * Dd] = o[r];
    }
  }
  __syncthreads();
  for (int i = threadIdx.x * 4; i < 32 * Nn; i += 256 * 4) {
    const int row = i >> 9, col = i & (Nn - 1);
    *(float4*)&weights[((long)(b * Nn + qBase + row)) * Nn + col] =
        *(float4*)&Wsum[row * P_LD + col];
  }
}

// xn1 = LN(xt + attn_out @ Wo + bo)
__global__ void oproj_ln_kernel(const float* __restrict__ ao,
                                const float* __restrict__ Wo,
                                const float* __restrict__ bo,
                                const float* __restrict__ x,
                                const float* __restrict__ g,
                                const float* __restrict__ bet,
                                float* __restrict__ xn1) {
  __shared__ __bf16 Asp[32 * AP_LD];
  __shared__ __bf16 Bsw[2 * BSW_PLANE];
  __shared__ float  Cs[32 * C_LD];
  const int rowBase = blockIdx.x * 32;
  const int wave = threadIdx.x >> 5, lane = threadIdx.x & 31;
  const int half = lane >> 4, nl = lane & 15;
  const int wrow = (wave >> 2) * 16, wcol = (wave & 3) * 64;
  load_a_panel<0, false>(ao, rowBase, Dd, Asp);
  v8f acc[4] = {};
  for (int k0 = 0; k0 < Dd; k0 += 32) {
    load_b_tile_sw(Wo, k0, 0, Dd, Bsw);
    __syncthreads();
    const v16bf a = load_a_frag(Asp + wrow * AP_LD + k0, AP_LD);
#pragma unroll
    for (int j = 0; j < 4; ++j)
      acc[j] = wmma_bf16(a, load_b_frag_sw(Bsw, (wcol >> 4) + j), acc[j]);
    __syncthreads();
  }
#pragma unroll
  for (int j = 0; j < 4; ++j) {
    const int c = wcol + 16 * j + nl;
    const float bv = bo[c];
    float* cp = Cs + (wrow + 8 * half) * C_LD + c;
#pragma unroll
    for (int r = 0; r < 8; ++r) cp[r * C_LD] = acc[j][r] + bv;
  }
  __syncthreads();
  {  // residual + LayerNorm
    const int t = threadIdx.x, row = t >> 3, seg = t & 7;
    const int gr = rowBase + row;
    const int b = gr >> 9, n = gr & (Nn - 1);
    const float* xr = x + ((long)n * Bb + b) * Dd + seg * 32;
    float v[32], s = 0.f, ss = 0.f;
#pragma unroll
    for (int i = 0; i < 32; ++i) {
      v[i] = Cs[row * C_LD + seg * 32 + i] + xr[i];
      s += v[i]; ss += v[i] * v[i];
    }
#pragma unroll
    for (int m = 1; m < 8; m <<= 1) { s += __shfl_xor(s, m, 32); ss += __shfl_xor(ss, m, 32); }
    const float mean = s * (1.0f / Dd);
    const float rstd = rsqrtf(ss * (1.0f / Dd) - mean * mean + 1e-5f);
    float* orow = xn1 + (long)gr * Dd + seg * 32;
#pragma unroll
    for (int i = 0; i < 32; ++i) {
      const int c = seg * 32 + i;
      orow[i] = (v[i] - mean) * rstd * g[c] + bet[c];
    }
  }
}

// e2a = LN(e + w * gathered) -> bf16 ; one wave per edge row
__global__ void edge_pre_kernel(const float* __restrict__ e,
                                const int* __restrict__ eidx,
                                const float* __restrict__ weights,
                                const float* __restrict__ x,
                                const float* __restrict__ g,
                                const float* __restrict__ bet,
                                __bf16* __restrict__ e2a) {
  const int wave = threadIdx.x >> 5, lane = threadIdx.x & 31;
  const int row = blockIdx.x * 8 + wave;
  const int b = row >> 13, n = (row >> 4) & (Nn - 1);
  const int idx = eidx[row];
  const float w = weights[((long)(b * Nn + idx)) * Nn + n];
  const float* er = e + (long)row * Dd;
  const float* xr = x + ((long)idx * Bb + b) * Dd;
  float v[8], s = 0.f, ss = 0.f;
#pragma unroll
  for (int i = 0; i < 8; ++i) {
    const int c = lane + 32 * i;
    v[i] = er[c] + w * xr[c];
    s += v[i]; ss += v[i] * v[i];
  }
#pragma unroll
  for (int m = 16; m; m >>= 1) { s += __shfl_xor(s, m, 32); ss += __shfl_xor(ss, m, 32); }
  const float mean = s * (1.0f / Dd);
  const float rstd = rsqrtf(ss * (1.0f / Dd) - mean * mean + 1e-5f);
#pragma unroll
  for (int i = 0; i < 8; ++i) {
    const int c = lane + 32 * i;
    e2a[(long)row * Dd + c] = (__bf16)((v[i] - mean) * rstd * g[c] + bet[c]);
  }
}

// out = LN(a + gelu(a@W1+b1)@W2 + b2), hidden kept in LDS (fully fused)
// OUTMODE 0: node path, write (N,B,D)-transposed. OUTMODE 1: edge, flat.
template <bool ABF16, int OUTMODE>
__global__ void fused_mlp_kernel(const void* __restrict__ Ain,
                                 const float* __restrict__ W1,
                                 const float* __restrict__ b1,
                                 const float* __restrict__ W2,
                                 const float* __restrict__ b2,
                                 const float* __restrict__ g,
                                 const float* __restrict__ bet,
                                 float* __restrict__ out) {
  __shared__ __bf16 Asp[32 * AP_LD];
  __shared__ __bf16 Bsw[2 * BSW_PLANE];
  __shared__ __bf16 Hs[32 * H_LD];
  __shared__ float  Cs[32 * C_LD];
  const int rowBase = blockIdx.x * 32;
  const int wave = threadIdx.x >> 5, lane = threadIdx.x & 31;
  const int half = lane >> 4, nl = lane & 15;
  const int wrow = (wave >> 2) * 16, wcol = (wave & 3) * 64;

  load_a_panel<0, ABF16>(Ain, rowBase, Dd, Asp);

  // Phase 1: H = gelu(A @ W1 + b1)
  for (int cb = 0; cb < 4; ++cb) {
    v8f acc[4] = {};
    for (int k0 = 0; k0 < Dd; k0 += 32) {
      load_b_tile_sw(W1, k0, cb * 256, Ff, Bsw);
      if (threadIdx.x < 32 && k0 + 32 < Dd)
        __builtin_prefetch(W1 + (long)(k0 + 32 + (int)threadIdx.x) * Ff + cb * 256, 0, 0);
      __syncthreads();
      const v16bf a = load_a_frag(Asp + wrow * AP_LD + k0, AP_LD);
#pragma unroll
      for (int j = 0; j < 4; ++j)
        acc[j] = wmma_bf16(a, load_b_frag_sw(Bsw, (wcol >> 4) + j), acc[j]);
      __syncthreads();
    }
#pragma unroll
    for (int j = 0; j < 4; ++j) {
      const int c = cb * 256 + wcol + 16 * j + nl;
      const float bv = b1[c];
      __bf16* hp = Hs + (wrow + 8 * half) * H_LD + c;
#pragma unroll
      for (int r = 0; r < 8; ++r) {
        const float v = acc[j][r] + bv;
        hp[r * H_LD] = (__bf16)(0.5f * v * (1.0f + erff(v * 0.70710678118654752f)));
      }
    }
  }
  __syncthreads();

  // Phase 2: C = H @ W2 + b2
  v8f acc[4] = {};
  for (int k0 = 0; k0 < Ff; k0 += 32) {
    load_b_tile_sw(W2, k0, 0, Dd, Bsw);
    if (threadIdx.x < 32 && k0 + 32 < Ff)
      __builtin_prefetch(W2 + (long)(k0 + 32 + (int)threadIdx.x) * Dd, 0, 0);
    __syncthreads();
    const v16bf a = load_a_frag(Hs + wrow * H_LD + k0, H_LD);
#pragma unroll
    for (int j = 0; j < 4; ++j)
      acc[j] = wmma_bf16(a, load_b_frag_sw(Bsw, (wcol >> 4) + j), acc[j]);
    __syncthreads();
  }
#pragma unroll
  for (int j = 0; j < 4; ++j) {
    const int c = wcol + 16 * j + nl;
    const float bv = b2[c];
    float* cp = Cs + (wrow + 8 * half) * C_LD + c;
#pragma unroll
    for (int r = 0; r < 8; ++r) cp[r * C_LD] = acc[j][r] + bv;
  }
  __syncthreads();

  // residual + LayerNorm + store
  {
    const int t = threadIdx.x, row = t >> 3, seg = t & 7;
    const long gr = rowBase + row;
    float v[32], s = 0.f, ss = 0.f;
#pragma unroll
    for (int i = 0; i < 32; ++i) {
      const int c = seg * 32 + i;
      const float res = ABF16 ? (float)((const __bf16*)Ain)[gr * Dd + c]
                              : ((const float*)Ain)[gr * Dd + c];
      v[i] = Cs[row * C_LD + c] + res;
      s += v[i]; ss += v[i] * v[i];
    }
#pragma unroll
    for (int m = 1; m < 8; m <<= 1) { s += __shfl_xor(s, m, 32); ss += __shfl_xor(ss, m, 32); }
    const float mean = s * (1.0f / Dd);
    const float rstd = rsqrtf(ss * (1.0f / Dd) - mean * mean + 1e-5f);
    float* orow;
    if (OUTMODE == 0) {
      const int b = (int)(gr >> 9), n = (int)(gr & (Nn - 1));
      orow = out + ((long)n * Bb + b) * Dd + seg * 32;
    } else {
      orow = out + gr * Dd + seg * 32;
    }
#pragma unroll
    for (int i = 0; i < 32; ++i) {
      const int c = seg * 32 + i;
      orow[i] = (v[i] - mean) * rstd * g[c] + bet[c];
    }
  }
}

// ---------------------------------------------------------------- launcher

extern "C" void kernel_launch(void* const* d_in, const int* in_sizes, int n_in,
                              void* d_out, int out_size, void* d_ws, size_t ws_size,
                              hipStream_t stream) {
  (void)in_sizes; (void)n_in; (void)out_size; (void)ws_size;
  const float* x    = (const float*)d_in[0];
  const float* e    = (const float*)d_in[1];
  const int*   eidx = (const int*)d_in[2];
  const float* Wq = (const float*)d_in[3];  const float* bq = (const float*)d_in[4];
  const float* Wk = (const float*)d_in[5];  const float* bk = (const float*)d_in[6];
  const float* Wv = (const float*)d_in[7];  const float* bv = (const float*)d_in[8];
  const float* Wo = (const float*)d_in[9];  const float* bo = (const float*)d_in[10];
  const float* nW1 = (const float*)d_in[11]; const float* nb1 = (const float*)d_in[12];
  const float* nW2 = (const float*)d_in[13]; const float* nb2 = (const float*)d_in[14];
  const float* eW1 = (const float*)d_in[15]; const float* eb1 = (const float*)d_in[16];
  const float* eW2 = (const float*)d_in[17]; const float* eb2 = (const float*)d_in[18];
  const float* na_g = (const float*)d_in[19]; const float* na_b = (const float*)d_in[20];
  const float* ea_g = (const float*)d_in[21]; const float* ea_b = (const float*)d_in[22];
  const float* nf_g = (const float*)d_in[23]; const float* nf_b = (const float*)d_in[24];
  const float* ef_g = (const float*)d_in[25]; const float* ef_b = (const float*)d_in[26];

  const size_t nnb = (size_t)Bb * Nn * Nn;
  float*  bias    = (float*)d_ws;
  float*  weights = bias + nnb;
  __bf16* qkv     = (__bf16*)(weights + nnb);
  const size_t perMat = (size_t)Bb * Hh * Nn * 32;
  __bf16* qb = qkv;
  __bf16* kb = qkv + perMat;
  __bf16* vb = qkv + 2 * perMat;
  float*  attn_out = (float*)(qkv + 3 * perMat);
  float*  xn1      = attn_out + (size_t)Bb * Nn * Dd;
  __bf16* e2a      = (__bf16*)(xn1 + (size_t)Bb * Nn * Dd);

  float* outN = (float*)d_out;                   // (N,B,D)
  float* outE = outN + (size_t)Nn * Bb * Dd;     // (B,N,K,D)

  zero_kernel<<<2048, 256, 0, stream>>>(bias, (long)nnb);
  scatter_bias_kernel<<<8192, 256, 0, stream>>>(e, eidx, bias);

  const float qscale = 0.17677669529663687f;  // 1/sqrt(32)
  qkv_gemm_kernel<<<128, 256, 0, stream>>>(x, Wq, bq, qb, qscale);
  qkv_gemm_kernel<<<128, 256, 0, stream>>>(x, Wk, bk, kb, 1.0f);
  qkv_gemm_kernel<<<128, 256, 0, stream>>>(x, Wv, bv, vb, 1.0f);

  attn_kernel<<<128, 256, 0, stream>>>(qb, kb, vb, bias, weights, attn_out);

  oproj_ln_kernel<<<128, 256, 0, stream>>>(attn_out, Wo, bo, x, na_g, na_b, xn1);

  edge_pre_kernel<<<8192, 256, 0, stream>>>(e, eidx, weights, x, ea_g, ea_b, e2a);

  fused_mlp_kernel<false, 0><<<128, 256, 0, stream>>>(xn1, nW1, nb1, nW2, nb2,
                                                      nf_g, nf_b, outN);
  fused_mlp_kernel<true, 1><<<2048, 256, 0, stream>>>(e2a, eW1, eb1, eW2, eb2,
                                                      ef_g, ef_b, outE);
}
